// TransducerModel_30786325578117
// MI455X (gfx1250) — compile-verified, matrix operand order checked by reading, not run on hardware
//
#include <hip/hip_runtime.h>
#include <hip/hip_bf16.h>
#include <math.h>

typedef __attribute__((ext_vector_type(16))) __bf16 v16bf;
typedef __attribute__((ext_vector_type(8)))  float  v8f;

#define B_   16
#define TMAX 256
#define U1   65     // U + 1
#define ENC_ 512
#define H3   1536   // 3 * 512
#define KT_  16     // 512 / 32 k-tiles (all GEMMs here have K = 512)
#define TILE_ELEMS 512   // one 32x16-ushort packed tile = 512 ushorts = 1024 B
#define VP   80     // V1=65 padded to 5 n-tiles

// ---------------- helpers ----------------

__device__ __forceinline__ unsigned short f2bf(float f) {
    union { float f; unsigned u; } v; v.f = f;
    unsigned u = v.u;
    u = (u + 0x7FFFu + ((u >> 16) & 1u)) >> 16;   // round-to-nearest-even
    return (unsigned short)u;
}

// WMMA 16-bit operand layout (A: row m, B: col n), k within 32-wide k-tile:
//   lane = mn + 16*((kk>>3)&1) ;  slot = (kk&7) + 8*((kk>>4)&1)
__device__ __forceinline__ int pk_lane(int mn, int kk) { return mn + (((kk >> 3) & 1) << 4); }
__device__ __forceinline__ int pk_slot(int kk)         { return (kk & 7) + (((kk >> 4) & 1) << 3); }

__device__ __forceinline__ float sigmoidf_(float x) { return 1.0f / (1.0f + expf(-x)); }

// ---------------- weight packing: f32 KxN row-major -> bf16 WMMA B-layout ----------------
// dst layout: [n_tile][k_tile][lane(32)][slot(16)]   (1024 B per 32x16 tile)

__global__ void pack_B_kernel(const float* __restrict__ src, int K, int Nsrc, int Npad,
                              unsigned short* __restrict__ dst) {
    int idx = blockIdx.x * blockDim.x + threadIdx.x;
    if (idx >= K * Npad) return;
    int k = idx / Npad, n = idx % Npad;
    float v = (n < Nsrc) ? src[(size_t)k * Nsrc + n] : 0.0f;
    int nt = n >> 4, nn = n & 15, kt = k >> 5, kk = k & 31;
    int Ktiles = K >> 5;
    dst[((size_t)(nt * Ktiles + kt) * 32 + pk_lane(nn, kk)) * 16 + pk_slot(kk)] = f2bf(v);
}

// ---------------- token index builders ----------------
__global__ void enc_tok_kernel(const int* __restrict__ x, int* __restrict__ tok) {
    int r = blockIdx.x * blockDim.x + threadIdx.x;
    if (r >= B_ * TMAX) return;
    int b = r & 15, t = r >> 4;
    tok[r] = x[b * TMAX + t];
}
__global__ void pred_tok_kernel(const int* __restrict__ y, int* __restrict__ tok) {
    int r = blockIdx.x * blockDim.x + threadIdx.x;
    if (r >= B_ * U1) return;
    int b = r & 15, u = r >> 4;
    tok[r] = (u == 0) ? 0 : y[b * 64 + (u - 1)];
}

// ---------------- embedding gather + pack to WMMA A-layout bf16 ----------------
__global__ void embed_pack_kernel(const int* __restrict__ tok, const float* __restrict__ emb,
                                  int M, unsigned short* __restrict__ dst) {
    int idx = blockIdx.x * blockDim.x + threadIdx.x;
    if (idx >= M * ENC_) return;
    int r = idx >> 9, k = idx & 511;
    float v = emb[(size_t)tok[r] * ENC_ + k];
    int mt = r >> 4, m = r & 15, kt = k >> 5, kk = k & 31;
    dst[((size_t)(mt * KT_ + kt) * 32 + pk_lane(m, kk)) * 16 + pk_slot(kk)] = f2bf(v);
}

// ---------------- WMMA GEMM (K=512 fixed), one wave per 16x16 tile, f32 out ----------------
__global__ void gemm_f32_kernel(const unsigned short* __restrict__ Apk,
                                const unsigned short* __restrict__ Bpk,
                                const float* __restrict__ bias,
                                int Mt, int Nt, int Npad, int Nbias,
                                float* __restrict__ out) {
    int lane = threadIdx.x & 31;
    int wv = (blockIdx.x * blockDim.x + threadIdx.x) >> 5;
    if (wv >= Mt * Nt) return;                 // wave-uniform exit (EXEC all-1 for WMMA)
    int mt = wv / Nt, nt = wv % Nt;
    // one base pointer per operand; K-tile stride is a compile-time immediate
    const unsigned short* ab = Apk + ((size_t)mt * KT_ * 32 + lane) * 16;
    const unsigned short* bb = Bpk + ((size_t)nt * KT_ * 32 + lane) * 16;
    v8f c = {0.f, 0.f, 0.f, 0.f, 0.f, 0.f, 0.f, 0.f};
#pragma unroll
    for (int kt = 0; kt < KT_; ++kt) {
        v16bf a = *(const v16bf*)(ab + kt * TILE_ELEMS);
        v16bf b = *(const v16bf*)(bb + kt * TILE_ELEMS);
        c = __builtin_amdgcn_wmma_f32_16x16x32_bf16(false, a, false, b, (short)0, c, false, false);
    }
    int nn = lane & 15, hi = lane >> 4;
    int col = nt * 16 + nn;
    float bv = (bias && col < Nbias) ? bias[col] : 0.0f;
#pragma unroll
    for (int r = 0; r < 8; ++r) {
        int row = mt * 16 + r + 8 * hi;        // C layout: lanes16-31 hold M = 8 + vgpr
        out[(size_t)row * Npad + col] = c[r] + bv;
    }
}

// ---------------- WMMA GEMM writing bf16 A-packed output (N multiple of 32) ----------------
__global__ void gemm_packA_kernel(const unsigned short* __restrict__ Apk,
                                  const unsigned short* __restrict__ Bpk,
                                  const float* __restrict__ bias,
                                  int Mt, int Nt,
                                  unsigned short* __restrict__ out) {
    int lane = threadIdx.x & 31;
    int wv = (blockIdx.x * blockDim.x + threadIdx.x) >> 5;
    if (wv >= Mt * Nt) return;
    int mt = wv / Nt, nt = wv % Nt;
    const unsigned short* ab = Apk + ((size_t)mt * KT_ * 32 + lane) * 16;
    const unsigned short* bb = Bpk + ((size_t)nt * KT_ * 32 + lane) * 16;
    v8f c = {0.f, 0.f, 0.f, 0.f, 0.f, 0.f, 0.f, 0.f};
#pragma unroll
    for (int kt = 0; kt < KT_; ++kt) {
        v16bf a = *(const v16bf*)(ab + kt * TILE_ELEMS);
        v16bf b = *(const v16bf*)(bb + kt * TILE_ELEMS);
        c = __builtin_amdgcn_wmma_f32_16x16x32_bf16(false, a, false, b, (short)0, c, false, false);
    }
    int nn = lane & 15, hi = lane >> 4;
    int col = nt * 16 + nn;
    float bv = bias ? bias[col] : 0.0f;
    int KTo = Nt >> 1;                         // this GEMM's N = next GEMM's K
    int kto = col >> 5, kko = col & 31;
#pragma unroll
    for (int r = 0; r < 8; ++r) {
        int m = r + 8 * hi;
        out[((size_t)(mt * KTo + kto) * 32 + pk_lane(m, kko)) * 16 + pk_slot(kko)] = f2bf(c[r] + bv);
    }
}

// ---------------- fused recurrent GRU (2 blocks: encoder / predictor concurrently) ----
__global__ __launch_bounds__(1024, 1) void gru_kernel(
    int steps0, const float* xz0, const unsigned short* R0, const float* br0,
    const float* h00, int bc0, unsigned short* H0,
    int steps1, const float* xz1, const unsigned short* R1, const float* br1,
    const float* h01, int bc1, unsigned short* H1) {
    extern __shared__ char smem[];
    unsigned short* hA = (unsigned short*)smem;               // 16 KB: packed A (16 k-tiles)
    float* hz = (float*)(smem + 16 * 1024);                   // 96 KB: hz staging [16][1536]
    float* hf = (float*)(smem + 16 * 1024 + 96 * 1024);       // 32 KB: h f32 [16][512]

    int steps; const float* xz; const unsigned short* R; const float* br;
    const float* h0; int bc; unsigned short* Hout;
    if (blockIdx.x == 0) { steps = steps0; xz = xz0; R = R0; br = br0; h0 = h00; bc = bc0; Hout = H0; }
    else                 { steps = steps1; xz = xz1; R = R1; br = br1; h0 = h01; bc = bc1; Hout = H1; }

    int tid = threadIdx.x;
    int lane = tid & 31, wv = tid >> 5;

    { // init h
        int b = tid >> 6, j0 = (tid & 63) * 8;
        for (int jj = 0; jj < 8; ++jj) {
            int j = j0 + jj;
            float h = (bc == 2) ? 0.0f : (bc == 1 ? h0[j] : h0[b * ENC_ + j]);
            hf[b * ENC_ + j] = h;
            int kt = j >> 5, kk = j & 31;
            hA[(kt * 32 + pk_lane(b, kk)) * 16 + pk_slot(kk)] = f2bf(h);
        }
    }
    __syncthreads();

    int nt0 = wv, nt1 = wv + 32, nt2 = wv + 64;
    const unsigned short* rb0 = R + ((size_t)nt0 * KT_ * 32 + lane) * 16;
    const unsigned short* rb1 = R + ((size_t)nt1 * KT_ * 32 + lane) * 16;
    const unsigned short* rb2 = R + ((size_t)nt2 * KT_ * 32 + lane) * 16;
    const unsigned short* alds = &hA[lane * 16];
    int gb = tid >> 6, gj0 = (tid & 63) * 8;

    for (int t = 0; t < steps; ++t) {
        // prefetch next step's xz rows while this step's WMMAs run
        if (t + 1 < steps) {
            const float* nxt = xz + ((size_t)(t + 1) * B_ + gb) * H3 + gj0;
            __builtin_prefetch(nxt, 0, 0);
            __builtin_prefetch(nxt + ENC_, 0, 0);
            __builtin_prefetch(nxt + 2 * ENC_, 0, 0);
        }
        v8f a0 = {0.f,0.f,0.f,0.f,0.f,0.f,0.f,0.f};
        v8f a1 = a0, a2 = a0;
#pragma unroll
        for (int kt = 0; kt < KT_; ++kt) {
            v16bf a  = *(const v16bf*)(alds + kt * TILE_ELEMS);
            v16bf b0 = *(const v16bf*)(rb0 + kt * TILE_ELEMS);
            a0 = __builtin_amdgcn_wmma_f32_16x16x32_bf16(false, a, false, b0, (short)0, a0, false, false);
            v16bf b1 = *(const v16bf*)(rb1 + kt * TILE_ELEMS);
            a1 = __builtin_amdgcn_wmma_f32_16x16x32_bf16(false, a, false, b1, (short)0, a1, false, false);
            v16bf b2 = *(const v16bf*)(rb2 + kt * TILE_ELEMS);
            a2 = __builtin_amdgcn_wmma_f32_16x16x32_bf16(false, a, false, b2, (short)0, a2, false, false);
        }
        int nn = lane & 15, hi = lane >> 4;
#pragma unroll
        for (int r = 0; r < 8; ++r) {
            int m = r + 8 * hi;
            hz[m * H3 + nt0 * 16 + nn] = a0[r];
            hz[m * H3 + nt1 * 16 + nn] = a1[r];
            hz[m * H3 + nt2 * 16 + nn] = a2[r];
        }
        __syncthreads();
        { // gates
            const float* xzr = xz + ((size_t)t * B_ + gb) * H3;
            for (int jj = 0; jj < 8; ++jj) {
                int j = gj0 + jj;
                float xzz = xzr[j], xzg = xzr[ENC_ + j], xzh = xzr[2 * ENC_ + j];
                float hzz = hz[gb * H3 + j] + br[j];
                float hzr = hz[gb * H3 + ENC_ + j] + br[ENC_ + j];
                float hzh = hz[gb * H3 + 2 * ENC_ + j] + br[2 * ENC_ + j];
                float z  = sigmoidf_(xzz + hzz);
                float rg = sigmoidf_(xzg + hzr);
                float hh = tanhf(xzh + rg * hzh);
                float h  = hf[gb * ENC_ + j];
                float hn = z * h + (1.0f - z) * hh;
                hf[gb * ENC_ + j] = hn;
                unsigned short bv = f2bf(hn);
                int kt = j >> 5, kk = j & 31;
                int off = (kt * 32 + pk_lane(gb, kk)) * 16 + pk_slot(kk);
                hA[off] = bv;
                Hout[(size_t)t * 8192 + off] = bv;   // A-packed tile for row-tile t
            }
        }
        __syncthreads();
    }
}

// ---------------- joint log-softmax: one wave per (b,t,u) ----------------
__global__ void joint_lp_kernel(const float* __restrict__ encJ, const float* __restrict__ predJ,
                                const int* __restrict__ y,
                                float* __restrict__ lpb, float* __restrict__ lpy) {
    int lane = threadIdx.x & 31;
    long wv = (long)blockIdx.x * (blockDim.x >> 5) + (threadIdx.x >> 5);
    if (wv >= (long)B_ * TMAX * U1) return;
    int u = (int)(wv % U1);
    int t = (int)((wv / U1) % TMAX);
    int b = (int)(wv / ((long)U1 * TMAX));
    const float* ej = encJ + ((size_t)t * B_ + b) * VP;    // join_b already folded in
    const float* pj = predJ + ((size_t)u * B_ + b) * VP;
    float v0 = ej[lane] + pj[lane];                        // v = 0..31
    float v1 = ej[lane + 32] + pj[lane + 32];              // v = 32..63
    float v2 = (lane == 0) ? (ej[64] + pj[64]) : -3.0e38f; // v = 64
    float m = fmaxf(fmaxf(v0, v1), v2);
#pragma unroll
    for (int off = 16; off > 0; off >>= 1) m = fmaxf(m, __shfl_xor(m, off, 32));
    float s = expf(v0 - m) + expf(v1 - m) + ((lane == 0) ? expf(v2 - m) : 0.0f);
#pragma unroll
    for (int off = 16; off > 0; off >>= 1) s += __shfl_xor(s, off, 32);
    float logZ = m + logf(s);
    float lpy_v = 0.0f;
    if (u < 64) {
        int yv = y[b * 64 + u];                            // 1..63
        float src = (yv < 32) ? __shfl(v0, yv, 32) : __shfl(v1, yv - 32, 32);
        lpy_v = src - logZ;
    }
    if (lane == 0) {
        lpb[((size_t)b * TMAX + t) * U1 + u] = v0 - logZ;  // v0 at lane 0 is v==NULL
        if (u < 64) lpy[((size_t)b * TMAX + t) * 64 + u] = lpy_v;
    }
}

// ---------------- RNNT alpha DP: 16 independent chains ----------------
__global__ void rnnt_dp_kernel(const float* __restrict__ lpb, const float* __restrict__ lpy,
                               const int* __restrict__ T, const int* __restrict__ U,
                               float* __restrict__ out) {
    __shared__ float al[B_ * U1];
    int b = threadIdx.x;
    if (b >= B_) return;
    float* a = &al[b * U1];
    const float* pb = lpb + (size_t)b * TMAX * U1;
    const float* py = lpy + (size_t)b * TMAX * 64;
    int Tb = T[b], Ub = U[b];
    a[0] = 0.0f;
    float c = 0.0f;
    for (int u = 1; u < U1; ++u) { c += py[u - 1]; a[u] = c; }   // alpha row t=0
    float resA = a[Ub];
    for (int t = 1; t < Tb; ++t) {
        const float* pbm = pb + (size_t)(t - 1) * U1;
        const float* pyt = py + (size_t)t * 64;
        float cprev = a[0] + pbm[0];
        a[0] = cprev;
        for (int u = 1; u < U1; ++u) {
            float fbu  = a[u] + pbm[u];                   // old row = alpha[t-1][u]
            float cand = cprev + pyt[u - 1];
            float mx = fmaxf(fbu, cand), mn = fminf(fbu, cand);
            float v = mx + log1pf(expf(mn - mx));         // logaddexp
            a[u] = v; cprev = v;
        }
        resA = a[Ub];
    }
    out[b] = -(resA + pb[(size_t)(Tb - 1) * U1 + Ub]);
}

// ---------------- host ----------------

extern "C" void kernel_launch(void* const* d_in, const int* in_sizes, int n_in,
                              void* d_out, int out_size, void* d_ws, size_t ws_size,
                              hipStream_t stream) {
    (void)in_sizes; (void)n_in; (void)out_size; (void)ws_size;
    const int*   x        = (const int*)  d_in[0];
    const int*   y        = (const int*)  d_in[1];
    const int*   Tt       = (const int*)  d_in[2];
    const int*   Uu       = (const int*)  d_in[3];
    const float* enc_emb  = (const float*)d_in[4];
    const float* enc_W    = (const float*)d_in[5];
    const float* enc_R    = (const float*)d_in[6];
    const float* enc_bi   = (const float*)d_in[7];
    const float* enc_br   = (const float*)d_in[8];
    const float* enc_lW   = (const float*)d_in[9];
    const float* enc_lb   = (const float*)d_in[10];
    const float* pred_emb = (const float*)d_in[11];
    const float* pred_W   = (const float*)d_in[12];
    const float* pred_R   = (const float*)d_in[13];
    const float* pred_bi  = (const float*)d_in[14];
    const float* pred_br  = (const float*)d_in[15];
    const float* pred_lW  = (const float*)d_in[16];
    const float* pred_lb  = (const float*)d_in[17];
    const float* pred_h0  = (const float*)d_in[18];
    const float* join_W   = (const float*)d_in[19];
    const float* join_b   = (const float*)d_in[20];
    float* out = (float*)d_out;

    char* p = (char*)d_ws;
    auto alloc = [&](size_t bytes) { void* r = (void*)p; p += (bytes + 255) & ~(size_t)255; return r; };

    unsigned short* encW_p   = (unsigned short*)alloc((size_t)1536 * 512 * 2);
    unsigned short* encR_p   = (unsigned short*)alloc((size_t)1536 * 512 * 2);
    unsigned short* predW_p  = (unsigned short*)alloc((size_t)1536 * 512 * 2);
    unsigned short* predR_p  = (unsigned short*)alloc((size_t)1536 * 512 * 2);
    unsigned short* enclW_p  = (unsigned short*)alloc((size_t)512 * 512 * 2);
    unsigned short* predlW_p = (unsigned short*)alloc((size_t)512 * 512 * 2);
    unsigned short* jenc_p   = (unsigned short*)alloc((size_t)80 * 512 * 2);
    unsigned short* jpred_p  = (unsigned short*)alloc((size_t)80 * 512 * 2);
    unsigned short* XencP    = (unsigned short*)alloc((size_t)4096 * 512 * 2);
    unsigned short* XpredP   = (unsigned short*)alloc((size_t)1040 * 512 * 2);
    unsigned short* HE       = (unsigned short*)alloc((size_t)256 * 8192 * 2);
    unsigned short* HP       = (unsigned short*)alloc((size_t)65 * 8192 * 2);
    unsigned short* encoutP  = (unsigned short*)alloc((size_t)4096 * 512 * 2);
    unsigned short* predoutP = (unsigned short*)alloc((size_t)1040 * 512 * 2);
    float* xzE   = (float*)alloc((size_t)4096 * 1536 * 4);
    float* xzP   = (float*)alloc((size_t)1040 * 1536 * 4);
    float* encJ  = (float*)alloc((size_t)4096 * 80 * 4);
    float* predJ = (float*)alloc((size_t)1040 * 80 * 4);
    float* lpb   = (float*)alloc((size_t)16 * 256 * 65 * 4);
    float* lpy   = (float*)alloc((size_t)16 * 256 * 64 * 4);
    int* tokE = (int*)alloc((size_t)4096 * 4);
    int* tokP = (int*)alloc((size_t)1040 * 4);

    // 1) pack weights to WMMA B layout (bf16)
    pack_B_kernel<<<3072, 256, 0, stream>>>(enc_W,  512, 1536, 1536, encW_p);
    pack_B_kernel<<<3072, 256, 0, stream>>>(enc_R,  512, 1536, 1536, encR_p);
    pack_B_kernel<<<3072, 256, 0, stream>>>(pred_W, 512, 1536, 1536, predW_p);
    pack_B_kernel<<<3072, 256, 0, stream>>>(pred_R, 512, 1536, 1536, predR_p);
    pack_B_kernel<<<1024, 256, 0, stream>>>(enc_lW,  512, 512, 512, enclW_p);
    pack_B_kernel<<<1024, 256, 0, stream>>>(pred_lW, 512, 512, 512, predlW_p);
    pack_B_kernel<<<160, 256, 0, stream>>>(join_W,            512, 65, 80, jenc_p);
    pack_B_kernel<<<160, 256, 0, stream>>>(join_W + 512 * 65, 512, 65, 80, jpred_p);

    // 2) embeddings -> bf16 A-packed
    enc_tok_kernel<<<16, 256, 0, stream>>>(x, tokE);
    pred_tok_kernel<<<5, 256, 0, stream>>>(y, tokP);
    embed_pack_kernel<<<8192, 256, 0, stream>>>(tokE, enc_emb, 4096, XencP);
    embed_pack_kernel<<<2080, 256, 0, stream>>>(tokP, pred_emb, 1040, XpredP);

    // 3) precompute xz = X @ W + bi  (non-recurrent half of GRU input)
    gemm_f32_kernel<<<3072, 256, 0, stream>>>(XencP,  encW_p,  enc_bi,  256, 96, 1536, 1536, xzE);
    gemm_f32_kernel<<<780,  256, 0, stream>>>(XpredP, predW_p, pred_bi, 65,  96, 1536, 1536, xzP);

    // 4) recurrent GRUs, encoder (block 0) and predictor (block 1) concurrently
    gru_kernel<<<2, 1024, 147456, stream>>>(
        256, xzE, encR_p, enc_br, nullptr, 2, HE,
        65,  xzP, predR_p, pred_br, pred_h0, 1, HP);

    // 5) output projections H @ lW + lb -> bf16 A-packed
    gemm_packA_kernel<<<1024, 256, 0, stream>>>(HE, enclW_p,  enc_lb,  256, 32, encoutP);
    gemm_packA_kernel<<<260,  256, 0, stream>>>(HP, predlW_p, pred_lb, 65,  32, predoutP);

    // 6) joint half-projections (join_b folded into encoder half)
    gemm_f32_kernel<<<160, 256, 0, stream>>>(encoutP,  jenc_p,  join_b,  256, 5, 80, 65, encJ);
    gemm_f32_kernel<<<41,  256, 0, stream>>>(predoutP, jpred_p, nullptr, 65,  5, 80, 0,  predJ);

    // 7) log-softmax over V for every (b,t,u); 8) alpha DP + loss
    joint_lp_kernel<<<33280, 256, 0, stream>>>(encJ, predJ, y, lpb, lpy);
    rnnt_dp_kernel<<<1, 64, 0, stream>>>(lpb, lpy, Tt, Uu, out);
}
